// GQAttention_49185965474218
// MI455X (gfx1250) — matrix operand order repaired and checked by reading, not compiled, not run on hardware
//
#include <hip/hip_runtime.h>
#include <hip/hip_bf16.h>

// ---------------------------------------------------------------------------
// GQA attention with ALiBi + causal + segment mask for MI455X (gfx1250).
// All matmuls via v_wmma_f32_16x16x32_f16 (wave32 WMMA, fp32 accumulate).
// GEMM waves own 4 independent accumulator tiles and double-buffer the
// next K-step's fragments in registers so WMMAs never wait on fresh loads.
// ---------------------------------------------------------------------------

typedef _Float16 f16;
typedef __attribute__((ext_vector_type(16))) _Float16 v16h;
typedef __attribute__((ext_vector_type(8)))  _Float16 v8h;
typedef __attribute__((ext_vector_type(8)))  float    v8f;

constexpr int Bc = 2, Lc = 2048, Dc = 1024, Hc = 16, KVc = 4, HDc = 64;

// ---- WMMA fragment loaders (layouts per CDNA5 ISA 7.12.2) ------------------
// A (16-bit, 16x32 MxK): lane holds row (lane&15); lanes 0-15: K = 0..7 and
// 16..23; lanes 16-31: K = 8..15 and 24..31  -> two contiguous 8-elem runs.
__device__ __forceinline__ v16h load_a_frag(const f16* p, int row0, int ld, int k0) {
  int lane = threadIdx.x & 31;
  int half = lane >> 4;
  const f16* base = p + (size_t)(row0 + (lane & 15)) * ld + k0 + 8 * half;
  v8h lo = *(const v8h*)(base);
  v8h hi = *(const v8h*)(base + 16);
  v16h r;
#pragma unroll
  for (int i = 0; i < 8; ++i) { r[i] = lo[i]; r[i + 8] = hi[i]; }
  return r;
}

// B (16-bit, 32x16 KxN, column-major load): lane holds col (lane&15);
// lanes 0-15: K = 0..7 and 16..23; lanes 16-31: K = 8..15 and 24..31 (same as A).
// `p` is a [N, K] row-major matrix whose row n is column n of B.
__device__ __forceinline__ v16h load_b_frag(const f16* p, int col0, int ld, int k0) {
  int lane = threadIdx.x & 31;
  int half = lane >> 4;
  const f16* base = p + (size_t)(col0 + (lane & 15)) * ld + k0 + 8 * half;
  v8h lo = *(const v8h*)(base);
  v8h hi = *(const v8h*)(base + 16);
  v16h r;
#pragma unroll
  for (int i = 0; i < 8; ++i) { r[i] = lo[i]; r[i + 8] = hi[i]; }
  return r;
}

__device__ __forceinline__ v8f wmma_f16(v16h a, v16h b, v8f c) {
  return __builtin_amdgcn_wmma_f32_16x16x32_f16(false, a, false, b, (short)0, c,
                                                false, false);
}

// ---- conversion kernels ----------------------------------------------------
__global__ void k_cvt(const float* __restrict__ s, f16* __restrict__ d, int n) {
  int i = blockIdx.x * blockDim.x + threadIdx.x;
  int stride = gridDim.x * blockDim.x;
  for (; i < n; i += stride) d[i] = (f16)s[i];
}

// src [rows, cols] row-major fp32 -> dst [cols, rows] f16 (transpose)
__global__ void k_tcvt(const float* __restrict__ s, f16* __restrict__ d,
                       int rows, int cols) {
  int i = blockIdx.x * blockDim.x + threadIdx.x;
  int n = rows * cols;
  int stride = gridDim.x * blockDim.x;
  for (; i < n; i += stride) {
    int r = i / cols, c = i - r * cols;
    d[(size_t)c * rows + r] = (f16)s[i];
  }
}

// ---- QKV projection GEMM ---------------------------------------------------
// Each wave: 16 rows x 64 cols = 4 independent accumulators; block = 4 waves
// stacked over M (64x64 macro-tile per block). Register double-buffering:
// K-step k0+32 fragments are loaded before the k0 WMMA burst issues.
// C[m,n] = sum_k X[m,k] * Wt[n,k], scattered into per-head layout.
// vtrans=0: out[((b*nh+h)*L + l)*HD + hd]   (Q, K)
// vtrans=1: out[((b*nh+h)*HD + hd)*L + l]   (V transposed for PV B-operand)
__global__ void __launch_bounds__(128)
k_gemm_proj(const f16* __restrict__ X, const f16* __restrict__ Wt,
            f16* __restrict__ out, int Kd, int nh, float scale, int vtrans) {
  int wave = threadIdx.x >> 5;
  int m0 = blockIdx.y * 64 + wave * 16;
  int n0 = blockIdx.x * 64;
  v8f acc[4];
#pragma unroll
  for (int t = 0; t < 4; ++t) { v8f z = {}; acc[t] = z; }

  v16h a  = load_a_frag(X, m0, Kd, 0);
  v16h b0 = load_b_frag(Wt, n0,      Kd, 0);
  v16h b1 = load_b_frag(Wt, n0 + 16, Kd, 0);
  v16h b2 = load_b_frag(Wt, n0 + 32, Kd, 0);
  v16h b3 = load_b_frag(Wt, n0 + 48, Kd, 0);

  for (int k0 = 0; k0 < Kd; k0 += 32) {
    int kn = k0 + 32;
    if (k0 + 64 < Kd) {  // two steps ahead (global_prefetch_b8)
      __builtin_prefetch(X + (size_t)m0 * Kd + k0 + 64, 0, 3);
      __builtin_prefetch(Wt + (size_t)n0 * Kd + k0 + 64, 0, 3);
    }
    v16h an = a, bn0 = b0, bn1 = b1, bn2 = b2, bn3 = b3;
    if (kn < Kd) {       // issue next step's loads before this step's WMMAs
      an  = load_a_frag(X, m0, Kd, kn);
      bn0 = load_b_frag(Wt, n0,      Kd, kn);
      bn1 = load_b_frag(Wt, n0 + 16, Kd, kn);
      bn2 = load_b_frag(Wt, n0 + 32, Kd, kn);
      bn3 = load_b_frag(Wt, n0 + 48, Kd, kn);
    }
    acc[0] = wmma_f16(a, b0, acc[0]);
    acc[1] = wmma_f16(a, b1, acc[1]);
    acc[2] = wmma_f16(a, b2, acc[2]);
    acc[3] = wmma_f16(a, b3, acc[3]);
    a = an; b0 = bn0; b1 = bn1; b2 = bn2; b3 = bn3;
  }

  int lane = threadIdx.x & 31;
  int half = lane >> 4, col = lane & 15;
#pragma unroll
  for (int t = 0; t < 4; ++t) {
#pragma unroll
    for (int r = 0; r < 8; ++r) {
      int m = m0 + r + 8 * half;        // global row = b*L + l
      int n = n0 + 16 * t + col;        // global col = h*HD + hd
      int bb = m >> 11, l = m & (Lc - 1);
      int h = n >> 6, hd = n & (HDc - 1);
      float v = acc[t][r] * scale;
      size_t o = vtrans ? (((size_t)(bb * nh + h) * HDc + hd) * Lc + l)
                        : (((size_t)(bb * nh + h) * Lc + l) * HDc + hd);
      out[o] = (f16)v;
    }
  }
}

// ---- flash attention: 1 wave per (b, h, 16-query tile) ---------------------
__global__ void __launch_bounds__(32)
k_attn(const f16* __restrict__ Q, const f16* __restrict__ K,
       const f16* __restrict__ Vt, const int* __restrict__ seg,
       f16* __restrict__ AO) {
  int qt = blockIdx.x, h = blockIdx.y, b = blockIdx.z;
  int kv = h >> 2;                      // G = H/KV = 4
  int q0 = qt * 16;
  int lane = threadIdx.x & 31, half = lane >> 4, col = lane & 15;
  float slope = exp2f(-0.5f * (float)(h + 1));  // alibi: base = 2^-0.5

  const f16* Qp = Q + (size_t)(b * Hc + h) * Lc * HDc;    // [L, HD]
  const f16* Kp = K + (size_t)(b * KVc + kv) * Lc * HDc;  // [L, HD]
  const f16* Vp = Vt + (size_t)(b * KVc + kv) * HDc * Lc; // [HD, L]
  const int* sb = seg + b * Lc;

  // Q tile loaded once (pre-scaled by 1/sqrt(HD) at projection time)
  v16h qa0 = load_a_frag(Qp, q0, HDc, 0);
  v16h qa1 = load_a_frag(Qp, q0, HDc, 32);

  int seg_q[8];
#pragma unroll
  for (int r = 0; r < 8; ++r) seg_q[r] = sb[q0 + r + 8 * half];

  float mrow[8], lrow[8];
  v8f acc[4];
#pragma unroll
  for (int r = 0; r < 8; ++r) { mrow[r] = -1e30f; lrow[r] = 0.f; }
#pragma unroll
  for (int t = 0; t < 4; ++t) { v8f z = {}; acc[t] = z; }

  __shared__ f16 P[16 * 32];            // P re-layout buffer (C-layout -> A-layout)

  int nit = (qt >> 1) + 1;              // 32 keys/iter covers causal extent
  for (int it = 0; it < nit; ++it) {
    int kt0 = it * 32;
    if (it + 1 < nit) {  // hint next iteration's K rows / V lines into cache
      int ktn = kt0 + 32;
      __builtin_prefetch(Kp + (size_t)(ktn + lane) * HDc, 0, 3);
      __builtin_prefetch(Vp + (size_t)lane * Lc + ktn, 0, 3);
      __builtin_prefetch(Vp + (size_t)(lane + 32) * Lc + ktn, 0, 3);
    }
    // scores S = Q @ K^T for 32 keys (2 sub-tiles of 16); loads hoisted
    v16h kb00 = load_b_frag(Kp, kt0, HDc, 0);
    v16h kb01 = load_b_frag(Kp, kt0, HDc, 32);
    v16h kb10 = load_b_frag(Kp, kt0 + 16, HDc, 0);
    v16h kb11 = load_b_frag(Kp, kt0 + 16, HDc, 32);
    v8f s0 = {}, s1 = {};
    s0 = wmma_f16(qa0, kb00, s0);
    s1 = wmma_f16(qa0, kb10, s1);
    s0 = wmma_f16(qa1, kb01, s0);
    s1 = wmma_f16(qa1, kb11, s1);

    int j0 = kt0 + col, j1 = j0 + 16;
    int sk0 = sb[j0], sk1 = sb[j1];

#pragma unroll
    for (int r = 0; r < 8; ++r) {
      int i = q0 + r + 8 * half;
      bool v0 = (i >= j0) && (seg_q[r] == sk0) && (sk0 >= 0);
      bool v1 = (i >= j1) && (seg_q[r] == sk1) && (sk1 >= 0);
      float x0 = v0 ? (s0[r] - slope * (float)(i - j0)) : -1e30f;
      float x1 = v1 ? (s1[r] - slope * (float)(i - j1)) : -1e30f;
      // row max across the 16 lanes of this half (xor masks < 16 stay in-half)
      float mx = fmaxf(x0, x1);
#pragma unroll
      for (int msk = 8; msk >= 1; msk >>= 1)
        mx = fmaxf(mx, __shfl_xor(mx, msk, 32));
      float mnew = fmaxf(mrow[r], mx);
      float alpha = __expf(mrow[r] - mnew);
      float p0 = v0 ? __expf(x0 - mnew) : 0.f;
      float p1 = v1 ? __expf(x1 - mnew) : 0.f;
      float ps = p0 + p1;
#pragma unroll
      for (int msk = 8; msk >= 1; msk >>= 1) ps += __shfl_xor(ps, msk, 32);
      lrow[r] = lrow[r] * alpha + ps;
      mrow[r] = mnew;
#pragma unroll
      for (int t = 0; t < 4; ++t) acc[t][r] *= alpha;
      P[(r + 8 * half) * 32 + col] = (f16)p0;
      P[(r + 8 * half) * 32 + col + 16] = (f16)p1;
    }
    __syncthreads();
    // O += P @ V over 4 HD n-tiles; loads hoisted ahead of the WMMA burst
    v16h pa  = load_a_frag(P, 0, 32, 0);    // P as A-fragment [16 x 32 keys]
    v16h vb0 = load_b_frag(Vp, 0,  Lc, kt0);
    v16h vb1 = load_b_frag(Vp, 16, Lc, kt0);
    v16h vb2 = load_b_frag(Vp, 32, Lc, kt0);
    v16h vb3 = load_b_frag(Vp, 48, Lc, kt0);
    acc[0] = wmma_f16(pa, vb0, acc[0]);
    acc[1] = wmma_f16(pa, vb1, acc[1]);
    acc[2] = wmma_f16(pa, vb2, acc[2]);
    acc[3] = wmma_f16(pa, vb3, acc[3]);
    __syncthreads();
  }

#pragma unroll
  for (int r = 0; r < 8; ++r) {
    float inv = 1.0f / lrow[r];
    int row = q0 + r + 8 * half;
#pragma unroll
    for (int t = 0; t < 4; ++t) {
      size_t o = (size_t)(b * Lc + row) * (Hc * HDc) + h * HDc + t * 16 + col;
      AO[o] = (f16)(acc[t][r] * inv);
    }
  }
}

// ---- output projection: out(fp32) = AO @ Wo --------------------------------
__global__ void __launch_bounds__(128)
k_gemm_out(const f16* __restrict__ X, const f16* __restrict__ Wt,
           float* __restrict__ out, int Kd) {
  int wave = threadIdx.x >> 5;
  int m0 = blockIdx.y * 64 + wave * 16;
  int n0 = blockIdx.x * 64;
  v8f acc[4];
#pragma unroll
  for (int t = 0; t < 4; ++t) { v8f z = {}; acc[t] = z; }

  v16h a  = load_a_frag(X, m0, Kd, 0);
  v16h b0 = load_b_frag(Wt, n0,      Kd, 0);
  v16h b1 = load_b_frag(Wt, n0 + 16, Kd, 0);
  v16h b2 = load_b_frag(Wt, n0 + 32, Kd, 0);
  v16h b3 = load_b_frag(Wt, n0 + 48, Kd, 0);

  for (int k0 = 0; k0 < Kd; k0 += 32) {
    int kn = k0 + 32;
    if (k0 + 64 < Kd) {
      __builtin_prefetch(X + (size_t)m0 * Kd + k0 + 64, 0, 3);
      __builtin_prefetch(Wt + (size_t)n0 * Kd + k0 + 64, 0, 3);
    }
    v16h an = a, bn0 = b0, bn1 = b1, bn2 = b2, bn3 = b3;
    if (kn < Kd) {
      an  = load_a_frag(X, m0, Kd, kn);
      bn0 = load_b_frag(Wt, n0,      Kd, kn);
      bn1 = load_b_frag(Wt, n0 + 16, Kd, kn);
      bn2 = load_b_frag(Wt, n0 + 32, Kd, kn);
      bn3 = load_b_frag(Wt, n0 + 48, Kd, kn);
    }
    acc[0] = wmma_f16(a, b0, acc[0]);
    acc[1] = wmma_f16(a, b1, acc[1]);
    acc[2] = wmma_f16(a, b2, acc[2]);
    acc[3] = wmma_f16(a, b3, acc[3]);
    a = an; b0 = bn0; b1 = bn1; b2 = bn2; b3 = bn3;
  }

  int lane = threadIdx.x & 31, half = lane >> 4, col = lane & 15;
#pragma unroll
  for (int t = 0; t < 4; ++t)
#pragma unroll
    for (int r = 0; r < 8; ++r)
      out[(size_t)(m0 + r + 8 * half) * Dc + n0 + 16 * t + col] = acc[t][r];
}

// ---------------------------------------------------------------------------
extern "C" void kernel_launch(void* const* d_in, const int* in_sizes, int n_in,
                              void* d_out, int out_size, void* d_ws, size_t ws_size,
                              hipStream_t stream) {
  const float* hs = (const float*)d_in[0];
  const int*   seg = (const int*)d_in[1];
  const float* Wq = (const float*)d_in[2];
  const float* Wk = (const float*)d_in[3];
  const float* Wv = (const float*)d_in[4];
  const float* Wo = (const float*)d_in[5];
  float* out = (float*)d_out;

  char* ws = (char*)d_ws;
  size_t off = 0;
  auto alloc = [&](size_t elems) {
    f16* p = (f16*)(ws + off);
    off += elems * sizeof(f16);
    return p;
  };
  f16* hsb = alloc((size_t)Bc * Lc * Dc);         // hs in f16            8 MB
  f16* Wqt = alloc((size_t)Dc * Hc * HDc);        // Wq^T [1024, D]       2 MB
  f16* Wkt = alloc((size_t)Dc * KVc * HDc);       // Wk^T [256, D]
  f16* Wvt = alloc((size_t)Dc * KVc * HDc);       // Wv^T [256, D]
  f16* Wot = alloc((size_t)Hc * HDc * Dc);        // Wo^T [1024, 1024]
  f16* Qb  = alloc((size_t)Bc * Hc * Lc * HDc);   // Q  [B,H,L,HD] scaled 8 MB
  f16* Kb  = alloc((size_t)Bc * KVc * Lc * HDc);  // K  [B,KV,L,HD]
  f16* Vtb = alloc((size_t)Bc * KVc * HDc * Lc);  // V^T[B,KV,HD,L]
  f16* AO  = alloc((size_t)Bc * Lc * Hc * HDc);   // attn out [B*L, 1024] 8 MB
  (void)ws_size; (void)in_sizes; (void)n_in; (void)out_size;

  const float SCALE = 0.125f;  // 1/sqrt(HD=64)

  // 1) precision conversion + weight transposes
  k_cvt<<<1024, 256, 0, stream>>>(hs, hsb, Bc * Lc * Dc);
  k_tcvt<<<(Dc * Hc * HDc + 255) / 256, 256, 0, stream>>>(Wq, Wqt, Dc, Hc * HDc);
  k_tcvt<<<(Dc * KVc * HDc + 255) / 256, 256, 0, stream>>>(Wk, Wkt, Dc, KVc * HDc);
  k_tcvt<<<(Dc * KVc * HDc + 255) / 256, 256, 0, stream>>>(Wv, Wvt, Dc, KVc * HDc);
  k_tcvt<<<(Hc * HDc * Dc + 255) / 256, 256, 0, stream>>>(Wo, Wot, Hc * HDc, Dc);

  // 2) QKV projections (WMMA GEMMs, 64x64 macro-tiles, double-buffered)
  dim3 gq(Hc * HDc / 64, Bc * Lc / 64);
  k_gemm_proj<<<gq, 128, 0, stream>>>(hsb, Wqt, Qb, Dc, Hc, SCALE, 0);
  dim3 gkv(KVc * HDc / 64, Bc * Lc / 64);
  k_gemm_proj<<<gkv, 128, 0, stream>>>(hsb, Wkt, Kb, Dc, KVc, 1.0f, 0);
  k_gemm_proj<<<gkv, 128, 0, stream>>>(hsb, Wvt, Vtb, Dc, KVc, 1.0f, 1);

  // 3) flash attention (ALiBi + causal + segment mask)
  dim3 ga(Lc / 16, Hc, Bc);
  k_attn<<<ga, 32, 0, stream>>>(Qb, Kb, Vtb, seg, AO);

  // 4) output projection -> fp32
  dim3 go(Dc / 64, Bc * Lc / 64);
  k_gemm_out<<<go, 128, 0, stream>>>(AO, Wot, out, Hc * HDc);
}
